// Mamba_54838142435976
// MI455X (gfx1250) — matrix operand order repaired and checked by reading, not compiled, not run on hardware
//
#include <hip/hip_runtime.h>
#include <hip/hip_bf16.h>
#include <math.h>
#include <stdint.h>

// ---------------------------------------------------------------------------
// Mamba forward for MI455X (gfx1250, wave32, WMMA + TDM), round 4.
//
//   0) one-shot fp32 -> bf16 conversion of all GEMM operand matrices
//   1) in_proj  GEMM via TDM: tensor_load_to_lds double-buffers 256x32 bf16
//      weight tiles into LDS (TENSORcnt + workgroup barrier), waves read
//      B fragments with ds_load_b128 and issue v_wmma_f32_16x16x32_bf16.
//   2) causal depthwise conv (K=4) + SiLU -> u (f32 + bf16)
//   3) x_proj   GEMM (global-load WMMA): x_dbl f32 + bf16
//   4) dt_proj  GEMM + fused softplus(x + 2*dt_proj_b)
//   5) selective scan: 1 lane per (b,d), 16-wide state in VGPRs -> y (bf16)
//   6) out_proj GEMM -> d_out
//
// Toolchain note: this environment's __builtin_amdgcn_tensor_load_to_lds is
// the 6-arg (clang-23 / therock-10.0) form:
//   (uint32x4 g0, int32x8 g1, int32x4 g2, int32x4 g3, int32x8 g4, i32 cpol)
// ---------------------------------------------------------------------------

#define D_MODEL 1024
#define D_STATE 16
#define D_INNER 2048
#define DT_RANK 64
#define BATCH   2
#define SEQLEN  1024
#define NTOK    (BATCH * SEQLEN)        // 2048 rows for all GEMMs
#define XDBL_W  (DT_RANK + 2 * D_STATE) // 96

typedef __attribute__((ext_vector_type(8)))  float          v8f;
typedef __attribute__((ext_vector_type(16))) __bf16         v16bf;
typedef __attribute__((ext_vector_type(16))) unsigned short v16us;
typedef __attribute__((ext_vector_type(8)))  unsigned short v8us;
typedef __attribute__((ext_vector_type(4)))  unsigned short v4us;
typedef __attribute__((ext_vector_type(4)))  unsigned int   v4u;
typedef __attribute__((ext_vector_type(8)))  int            v8i;
typedef __attribute__((ext_vector_type(4)))  int            v4i;

__device__ __forceinline__ unsigned short f32_to_bf16_rne(float f) {
  unsigned int u = __float_as_uint(f);
  unsigned int r = 0x7FFFu + ((u >> 16) & 1u);
  return (unsigned short)((u + r) >> 16);
}

__device__ __forceinline__ v16bf join16(v8us lo, v8us hi) {
  v16us t = __builtin_shufflevector(lo, hi, 0, 1, 2, 3, 4, 5, 6, 7,
                                    8, 9, 10, 11, 12, 13, 14, 15);
  return __builtin_bit_cast(v16bf, t);
}

// Bulk fp32 -> bf16 (4 elems/thread; n must be a multiple of 1024).
__global__ void cvt_bf16_kernel(const float* __restrict__ in,
                                unsigned short* __restrict__ out) {
  const int i = blockIdx.x * blockDim.x + threadIdx.x;
  float4 v = ((const float4*)in)[i];
  v4us o;
  o[0] = f32_to_bf16_rne(v.x);
  o[1] = f32_to_bf16_rne(v.y);
  o[2] = f32_to_bf16_rne(v.z);
  o[3] = f32_to_bf16_rne(v.w);
  ((v4us*)out)[i] = o;
}

// ======================= TDM-staged GEMM (in_proj) =========================
// C[m,n] = sum_k A[m,k] * W[n,k].  One block = 16(M) x 256(N) macro tile,
// 8 waves, wave w owns n = nbase + 32w (two 16x16 WMMA accumulators).
// W tiles (256 rows x 32 bf16) are DMA'd into LDS by the Tensor Data Mover,
// double buffered, with TDM row padding: 64B rows padded to 80B
// (pad_interval=3 -> every 16 DWORDs, pad_amount=3 -> 4 DWORDs).

#define TDM_ROWS   256
#define TDM_KSTEP  32
#define TDM_ROWPAD 40   // shorts per LDS row (32 data + 8 pad) = 80 bytes

#if __has_builtin(__builtin_amdgcn_tensor_load_to_lds)
#define HAVE_TDM 1
#else
#define HAVE_TDM 0
#endif

__global__ void gemm_wmma_bf16_tdm(const unsigned short* __restrict__ A, int lda,
                                   const unsigned short* __restrict__ W, int ldw,
                                   float* __restrict__ C, int ldc,
                                   int M, int N, int K) {
  __shared__ unsigned short Bbuf[2][TDM_ROWS * TDM_ROWPAD]; // 2 x 20 KB

  const int wave = threadIdx.x >> 5;
  const int lane = threadIdx.x & 31;
  const int lm   = lane & 15;
  const int hi   = lane >> 4;

  const int nblocks = N >> 8;             // 256-wide macro tiles
  const int m0      = (blockIdx.x / nblocks) << 4;
  const int nbase   = (blockIdx.x % nblocks) << 8;
  const int n0      = nbase + (wave << 5);

  // A lane: one 16-elem contiguous run (k = 16*hi + s); identical across waves.
  const unsigned short* arow = A + (size_t)(m0 + lm) * lda + 16 * hi;

#if HAVE_TDM
  // D# group1 is loop invariant: data_size=2B, pad 80B rows, dims/strides.
  v8i g1;
  g1[0] = (1 << 16) | (1 << 20) | (3 << 22) | (3 << 25); // size|pad_en|intvl|amt
  g1[1] = (int)((unsigned)(ldw & 0xFFFF) << 16);         // tensor_dim0 lo16
  g1[2] = (int)(((unsigned)ldw >> 16) | ((unsigned)(N & 0xFFFF) << 16));
  g1[3] = (int)(((unsigned)N >> 16) | ((unsigned)TDM_KSTEP << 16)); // tile_dim0
  g1[4] = TDM_ROWS;                                      // tile_dim1 (tile_dim2=0)
  g1[5] = ldw;                                           // dim0_stride lo32
  g1[6] = 0;
  g1[7] = 0;
  v4i gz4 = 0;                                           // groups 2/3: 2D tile
  v8i gz8 = 0;                                           // extra group (6-arg form)
#endif

  v8f acc0 = 0.0f, acc1 = 0.0f;

  // Issue TDM load of W[nbase..+255][k0..k0+31] into Bbuf[buf].
  auto stage = [&](int k0, int buf) {
#if HAVE_TDM
    uint64_t ga  = (uint64_t)(uintptr_t)(W + (size_t)nbase * ldw + k0);
    unsigned lds = (unsigned)(uintptr_t)(&Bbuf[buf][0]);
    v4u g0;
    g0[0] = 1u;                                          // count=1, user mode
    g0[1] = lds;                                         // lds_addr (bytes)
    g0[2] = (unsigned)(ga & 0xFFFFFFFFu);                // global_addr lo
    g0[3] = (unsigned)((ga >> 32) & 0x1FFFFFFu) | (2u << 30); // hi | type=2
    __builtin_amdgcn_tensor_load_to_lds(g0, g1, gz4, gz4, gz8, 0);
#else
    // Fallback: cooperative copy, thread t copies row t (32 bf16 = 4x16B).
    const unsigned short* src = W + (size_t)(nbase + threadIdx.x) * ldw + k0;
    unsigned short* dst = &Bbuf[buf][threadIdx.x * TDM_ROWPAD];
    *(v8us*)(dst)      = *(const v8us*)(src);
    *(v8us*)(dst + 8)  = *(const v8us*)(src + 8);
    *(v8us*)(dst + 16) = *(const v8us*)(src + 16);
    *(v8us*)(dst + 24) = *(const v8us*)(src + 24);
#endif
  };

  // Prologue: stage K-step 0 into buffer 0.
#if HAVE_TDM
  if (wave == 0) {
    stage(0, 0);
    __builtin_amdgcn_s_wait_tensorcnt(0);
  }
#else
  stage(0, 0);
#endif
  __syncthreads();

  int cur = 0;
  for (int k0 = 0; k0 < K; k0 += TDM_KSTEP) {
    const bool more = (k0 + TDM_KSTEP) < K;
#if HAVE_TDM
    if (wave == 0 && more) stage(k0 + TDM_KSTEP, cur ^ 1); // overlap DMA
#else
    if (more) stage(k0 + TDM_KSTEP, cur ^ 1);
#endif

    v8us a_lo = *(const v8us*)(arow + k0);
    v8us a_hi = *(const v8us*)(arow + k0 + 8);
    // B fragments from LDS (ds_load_b128): row = n - nbase, col base 16*hi.
    const unsigned short* b0p = &Bbuf[cur][((wave << 5) + lm) * TDM_ROWPAD + 16 * hi];
    const unsigned short* b1p = b0p + 16 * TDM_ROWPAD;
    v8us b0l = *(const v8us*)(b0p);
    v8us b0h = *(const v8us*)(b0p + 8);
    v8us b1l = *(const v8us*)(b1p);
    v8us b1h = *(const v8us*)(b1p + 8);

    v16bf av  = join16(a_lo, a_hi);
    acc0 = __builtin_amdgcn_wmma_f32_16x16x32_bf16(
        false, av, false, join16(b0l, b0h), (short)0, acc0, false, false);
    acc1 = __builtin_amdgcn_wmma_f32_16x16x32_bf16(
        false, av, false, join16(b1l, b1h), (short)0, acc1, false, false);

#if HAVE_TDM
    if (wave == 0 && more) __builtin_amdgcn_s_wait_tensorcnt(0);
#endif
    __syncthreads();   // publish next buffer / retire reads of current
    cur ^= 1;
  }

  const int col0 = n0 + lm;
#pragma unroll
  for (int v = 0; v < 8; ++v) {
    const size_t r = (size_t)(m0 + v + 8 * hi) * ldc;
    C[r + col0]      = acc0[v];
    C[r + col0 + 16] = acc1[v];
  }
}

// =================== global-load GEMM (x_proj/dt/out) ======================
// MODE 0: store f32 C.  MODE 1: softplus(val + 2*bias[n]).  MODE 2: f32+bf16.
template<int MODE>
__global__ void gemm_wmma_bf16(const unsigned short* __restrict__ A, int lda,
                               const unsigned short* __restrict__ W, int ldw,
                               float* __restrict__ C, int ldc,
                               unsigned short* __restrict__ C2,
                               int M, int N, int K,
                               const float* __restrict__ bias) {
  const int wid      = (blockIdx.x * blockDim.x + threadIdx.x) >> 5;
  const int tiles_n2 = N >> 5;
  const int tiles_m  = M >> 4;
  if (wid >= tiles_m * tiles_n2) return;
  const int m0 = (wid / tiles_n2) << 4;
  const int n0 = (wid % tiles_n2) << 5;

  const int lane = threadIdx.x & 31;
  const int lm   = lane & 15;
  const int hi   = lane >> 4;

  v8f acc0 = 0.0f, acc1 = 0.0f;

  const unsigned short* arow = A + (size_t)(m0 + lm) * lda + 16 * hi;
  const unsigned short* w0 = W + (size_t)(n0 + lm) * ldw + 16 * hi;
  const unsigned short* w1 = W + (size_t)(n0 + 16 + lm) * ldw + 16 * hi;

  for (int k0 = 0; k0 < K; k0 += 32) {
    v8us a_lo = *(const v8us*)(arow + k0);
    v8us a_hi = *(const v8us*)(arow + k0 + 8);
    v8us b0l  = *(const v8us*)(w0 + k0);
    v8us b0h  = *(const v8us*)(w0 + k0 + 8);
    v8us b1l  = *(const v8us*)(w1 + k0);
    v8us b1h  = *(const v8us*)(w1 + k0 + 8);
    if (k0 + 32 < K) {
      __builtin_prefetch(arow + k0 + 32, 0, 1);
      __builtin_prefetch(w0 + k0 + 32, 0, 1);
      __builtin_prefetch(w1 + k0 + 32, 0, 1);
    }
    v16bf av  = join16(a_lo, a_hi);
    acc0 = __builtin_amdgcn_wmma_f32_16x16x32_bf16(
        false, av, false, join16(b0l, b0h), (short)0, acc0, false, false);
    acc1 = __builtin_amdgcn_wmma_f32_16x16x32_bf16(
        false, av, false, join16(b1l, b1h), (short)0, acc1, false, false);
  }

  const int col0 = n0 + lm;
  const int col1 = col0 + 16;
  const float bb0 = (MODE == 1) ? bias[col0] : 0.0f;
  const float bb1 = (MODE == 1) ? bias[col1] : 0.0f;
#pragma unroll
  for (int v = 0; v < 8; ++v) {
    const size_t r = (size_t)(m0 + v + 8 * hi) * ldc;
    float x0 = acc0[v], x1 = acc1[v];
    if (MODE == 1) {
      float t0 = x0 + 2.0f * bb0;                   // reference adds dt bias twice
      float t1 = x1 + 2.0f * bb1;
      x0 = (t0 > 20.0f) ? t0 : log1pf(__expf(t0));  // softplus
      x1 = (t1 > 20.0f) ? t1 : log1pf(__expf(t1));
    }
    C[r + col0] = x0;
    C[r + col1] = x1;
    if (MODE == 2) {
      C2[r + col0] = f32_to_bf16_rne(x0);
      C2[r + col1] = f32_to_bf16_rne(x1);
    }
  }
}

// Causal depthwise conv1d (K=4) + SiLU. x is xz[..., 0:2048].
__global__ void conv_silu_kernel(const float* __restrict__ xz,
                                 const float* __restrict__ cw,
                                 const float* __restrict__ cb,
                                 float* __restrict__ u,
                                 unsigned short* __restrict__ ubf) {
  const int idx = blockIdx.x * blockDim.x + threadIdx.x; // b<<21 | l<<11 | d
  const int d = idx & (D_INNER - 1);
  const int l = (idx >> 11) & (SEQLEN - 1);
  const int b = idx >> 21;
  float acc = cb[d];
#pragma unroll
  for (int j = 0; j < 4; ++j) {
    int lp = l - 3 + j;
    if (lp >= 0)
      acc += xz[((size_t)(b * SEQLEN + lp)) * (2 * D_INNER) + d] * cw[d * 4 + j];
  }
  const float s = acc / (1.0f + __expf(-acc));  // SiLU
  u[idx] = s;
  ubf[idx] = f32_to_bf16_rne(s);
}

// Selective scan + skip + gating. One thread per (b,d); 16 states in VGPRs.
__global__ void scan_kernel(const float* __restrict__ delta,
                            const float* __restrict__ u,
                            const float* __restrict__ xdbl,
                            const float* __restrict__ xz,
                            const float* __restrict__ A_log,
                            const float* __restrict__ Dp,
                            unsigned short* __restrict__ ybf) {
  const int t = blockIdx.x * blockDim.x + threadIdx.x; // 0..4095
  const int b = t >> 11;
  const int d = t & (D_INNER - 1);

  float Arow[D_STATE];
#pragma unroll
  for (int n = 0; n < D_STATE; ++n) Arow[n] = -__expf(A_log[d * D_STATE + n]);
  const float Dv = Dp[d];

  float st[D_STATE];
#pragma unroll
  for (int n = 0; n < D_STATE; ++n) st[n] = 0.0f;

  for (int l = 0; l < SEQLEN; ++l) {
    const size_t row = (size_t)(b * SEQLEN + l);
    const float dl = delta[row * D_INNER + d];
    const float uu = u[row * D_INNER + d];
    const float* xr = xdbl + row * XDBL_W;
    float4 B0 = *(const float4*)(xr + DT_RANK);
    float4 B1 = *(const float4*)(xr + DT_RANK + 4);
    float4 B2 = *(const float4*)(xr + DT_RANK + 8);
    float4 B3 = *(const float4*)(xr + DT_RANK + 12);
    float4 C0 = *(const float4*)(xr + DT_RANK + 16);
    float4 C1 = *(const float4*)(xr + DT_RANK + 20);
    float4 C2 = *(const float4*)(xr + DT_RANK + 24);
    float4 C3 = *(const float4*)(xr + DT_RANK + 28);
    float Bv[16] = {B0.x, B0.y, B0.z, B0.w, B1.x, B1.y, B1.z, B1.w,
                    B2.x, B2.y, B2.z, B2.w, B3.x, B3.y, B3.z, B3.w};
    float Cv[16] = {C0.x, C0.y, C0.z, C0.w, C1.x, C1.y, C1.z, C1.w,
                    C2.x, C2.y, C2.z, C2.w, C3.x, C3.y, C3.z, C3.w};
    float ydot = 0.0f;
#pragma unroll
    for (int n = 0; n < D_STATE; ++n) {
      float dA = __expf(dl * Arow[n]);
      st[n] = dA * st[n] + dl * Bv[n] * uu;
      ydot += st[n] * Cv[n];
    }
    const float zv = xz[row * (2 * D_INNER) + D_INNER + d];
    const float gate = zv / (1.0f + __expf(-zv));
    ybf[row * D_INNER + d] = f32_to_bf16_rne((ydot + uu * Dv) * gate);
  }
}

extern "C" void kernel_launch(void* const* d_in, const int* in_sizes, int n_in,
                              void* d_out, int out_size, void* d_ws, size_t ws_size,
                              hipStream_t stream) {
  const float* hs        = (const float*)d_in[0];
  const float* in_proj_w = (const float*)d_in[1];
  const float* conv_w    = (const float*)d_in[2];
  const float* conv_b    = (const float*)d_in[3];
  const float* x_proj_w  = (const float*)d_in[4];
  const float* dt_proj_w = (const float*)d_in[5];
  const float* dt_proj_b = (const float*)d_in[6];
  const float* A_log     = (const float*)d_in[7];
  const float* Dp        = (const float*)d_in[8];
  const float* out_proj_w= (const float*)d_in[9];
  float* out = (float*)d_out;

  // ---- workspace layout (bytes; every block is a multiple of 16B) ----
  char* p = (char*)d_ws;
  float* xz      = (float*)p;          p += (size_t)NTOK * 2 * D_INNER * 4; // 32 MB
  float* u       = (float*)p;          p += (size_t)NTOK * D_INNER * 4;     // 16 MB
  float* xdbl    = (float*)p;          p += (size_t)NTOK * XDBL_W * 4;      // .75 MB
  float* delta   = (float*)p;          p += (size_t)NTOK * D_INNER * 4;     // 16 MB
  unsigned short* hs_bf   = (unsigned short*)p; p += (size_t)NTOK * D_MODEL * 2;
  unsigned short* win_bf  = (unsigned short*)p; p += (size_t)2 * D_INNER * D_MODEL * 2;
  unsigned short* wxp_bf  = (unsigned short*)p; p += (size_t)XDBL_W * D_INNER * 2;
  unsigned short* wdt_bf  = (unsigned short*)p; p += (size_t)D_INNER * DT_RANK * 2;
  unsigned short* wout_bf = (unsigned short*)p; p += (size_t)D_MODEL * D_INNER * 2;
  unsigned short* u_bf    = (unsigned short*)p; p += (size_t)NTOK * D_INNER * 2;
  unsigned short* xdbl_bf = (unsigned short*)p; p += (size_t)NTOK * XDBL_W * 2;
  unsigned short* y_bf    = (unsigned short*)p; p += (size_t)NTOK * D_INNER * 2;

  const int TPB = 256; // 8 waves (wave32) per block

  // ---- 0) one-shot bf16 conversion of GEMM operands ----
  cvt_bf16_kernel<<<(NTOK * D_MODEL) / 4 / TPB, TPB, 0, stream>>>(hs, hs_bf);
  cvt_bf16_kernel<<<(2 * D_INNER * D_MODEL) / 4 / TPB, TPB, 0, stream>>>(in_proj_w, win_bf);
  cvt_bf16_kernel<<<(XDBL_W * D_INNER) / 4 / TPB, TPB, 0, stream>>>(x_proj_w, wxp_bf);
  cvt_bf16_kernel<<<(D_INNER * DT_RANK) / 4 / TPB, TPB, 0, stream>>>(dt_proj_w, wdt_bf);
  cvt_bf16_kernel<<<(D_MODEL * D_INNER) / 4 / TPB, TPB, 0, stream>>>(out_proj_w, wout_bf);

  // ---- 1) in_proj via TDM-staged GEMM: M=2048 N=4096 K=1024 ----
  // blocks = (M/16) * (N/256) = 128 * 16
  gemm_wmma_bf16_tdm<<<128 * 16, TPB, 0, stream>>>(
      hs_bf, D_MODEL, win_bf, D_MODEL, xz, 2 * D_INNER,
      NTOK, 2 * D_INNER, D_MODEL);

  // ---- 2) depthwise conv + SiLU ----
  conv_silu_kernel<<<(BATCH * SEQLEN * D_INNER) / TPB, TPB, 0, stream>>>(
      xz, conv_w, conv_b, u, u_bf);

  // ---- 3) x_proj: x_dbl = u @ x_proj_w^T   M=2048 N=96 K=2048 ----
  gemm_wmma_bf16<2><<<(128 * 3) / 8, TPB, 0, stream>>>(
      u_bf, D_INNER, wxp_bf, D_INNER, xdbl, XDBL_W, xdbl_bf,
      NTOK, XDBL_W, D_INNER, nullptr);

  // ---- 4) dt_proj + fused softplus(x + 2b)  M=2048 N=2048 K=64 ----
  gemm_wmma_bf16<1><<<(128 * 64) / 8, TPB, 0, stream>>>(
      xdbl_bf, XDBL_W, wdt_bf, DT_RANK, delta, D_INNER, nullptr,
      NTOK, D_INNER, DT_RANK, dt_proj_b);

  // ---- 5) selective scan + skip + gating -> y (bf16) ----
  scan_kernel<<<(BATCH * D_INNER) / TPB, TPB, 0, stream>>>(
      delta, u, xdbl, xz, A_log, Dp, y_bf);

  // ---- 6) out_proj: out = y @ out_proj_w^T  M=2048 N=1024 K=2048 ----
  gemm_wmma_bf16<0><<<(128 * 32) / 8, TPB, 0, stream>>>(
      y_bf, D_INNER, wout_bf, D_INNER, out, D_MODEL, nullptr,
      NTOK, D_MODEL, D_INNER, nullptr);
}
